// LightGNN_72198400246405
// MI455X (gfx1250) — matrix operand order repaired and checked by reading, not compiled, not run on hardware
//
#include <hip/hip_runtime.h>
#include <hip/hip_bf16.h>

#define N_NODES 30000
#define N_EDGES 480000
#define N_GRAPH 32
#define DIN     64
#define HDIM    128
#define NLAYERS 4
#define GN_EPS  1e-5f

typedef __attribute__((ext_vector_type(16))) __bf16 v16bf;
typedef __attribute__((ext_vector_type(8)))  float  v8f;

__device__ __forceinline__ void atomic_add_f32(float* p, float v) {
    (void)__hip_atomic_fetch_add(p, v, __ATOMIC_RELAXED, __HIP_MEMORY_SCOPE_AGENT);
}

// ---------------------------------------------------------------------------
// Tiled GEMM: C[M x NOUT] = act(A[M x KD] @ W[KD x NOUT] + bias)
// bf16 WMMA (v_wmma_f32_16x16x32_bf16), f32 accumulate.
// 128 threads = 4 waves, each wave computes 16 rows x NOUT cols.
// Weights staged transposed in LDS as bf16 so B-fragment reads are contiguous.
// ---------------------------------------------------------------------------
template <int ACT, int KD, int NOUT>
__global__ __launch_bounds__(128) void gemm_bf16_wmma(
    const float* __restrict__ A, const float* __restrict__ W,
    const float* __restrict__ bias, float* __restrict__ C, int M) {
    constexpr int NT = NOUT / 16;
    __shared__ __bf16 sW[NOUT * KD];   // transposed: sW[n*KD + k]
    __shared__ float  sB[NOUT];

    const int tid = threadIdx.x;
    for (int idx = tid; idx < KD * NOUT; idx += 128) {
        const int k = idx / NOUT, n = idx % NOUT;
        sW[n * KD + k] = (__bf16)W[idx];
    }
    for (int i = tid; i < NOUT; i += 128) sB[i] = bias ? bias[i] : 0.0f;
    __syncthreads();

    const int wave = tid >> 5, lane = tid & 31;
    const int hi   = lane >> 4;     // 0 => lanes 0-15, 1 => lanes 16-31
    const int lrow = lane & 15;
    const int rowBase = blockIdx.x * 64 + wave * 16;
    const int row  = rowBase + lrow;
    const int rowc = row < M ? row : (M - 1);   // clamp (WMMA needs full EXEC)
    const float* Arow = A + (size_t)rowc * KD;

    v8f acc[NT] = {};
#pragma unroll
    for (int kb = 0; kb < KD; kb += 32) {
        // A fragment (contiguous k order, matching B):
        //   lanes 0-15: elems 0-15 = K+0..15
        //   lanes16-31: elems 0-15 = K+16..31
        v16bf a;
#pragma unroll
        for (int j = 0; j < 16; ++j) a[j] = (__bf16)Arow[kb + hi * 16 + j];

#pragma unroll
        for (int t = 0; t < NT; ++t) {
            // B fragment 32x16: lanes 0-15 (n=lane): K+0..15; lanes16-31: K+16..31
            const __bf16* wt = &sW[(t * 16 + lrow) * KD + kb + hi * 16];
            v16bf b;
#pragma unroll
            for (int j = 0; j < 16; ++j) b[j] = wt[j];
            acc[t] = __builtin_amdgcn_wmma_f32_16x16x32_bf16(
                false, a, false, b, (short)0, acc[t], false, false);
        }
    }

    // C/D f32 layout: VGPR e -> row = rowBase + e + hi*8, col = lrow + t*16
#pragma unroll
    for (int t = 0; t < NT; ++t) {
        const int   c  = t * 16 + lrow;
        const float bb = sB[c];
#pragma unroll
        for (int e = 0; e < 8; ++e) {
            const int r = rowBase + hi * 8 + e;
            if (r < M) {
                float v = acc[t][e] + bb;
                if (ACT == 1) v = v > 0.f ? v : 0.f;
                C[(size_t)r * NOUT + c] = v;
            }
        }
    }
}

// ---------------------------------------------------------------------------
// Small elementwise / reduction kernels
// ---------------------------------------------------------------------------
__global__ void zero_f32(float* p, int n) {
    int i = blockIdx.x * blockDim.x + threadIdx.x;
    if (i < n) p[i] = 0.f;
}

__global__ void count_nodes(const int* __restrict__ batch, float* __restrict__ gcnt) {
    int i = blockIdx.x * blockDim.x + threadIdx.x;
    if (i < N_NODES) atomic_add_f32(&gcnt[batch[i]], 1.0f);
}

__global__ void deg_init(float* deg) {   // self loop fill value 2.0 (improved=True)
    int i = blockIdx.x * blockDim.x + threadIdx.x;
    if (i < N_NODES) deg[i] = 2.0f;
}

__global__ void deg_accum(const int* __restrict__ ei, const float* __restrict__ ewp,
                          float* __restrict__ deg) {
    int e = blockIdx.x * blockDim.x + threadIdx.x;
    if (e < N_EDGES) {
        const float w = ewp[0];
        atomic_add_f32(&deg[ei[e]], w);             // col = e0 (flipped)
        atomic_add_f32(&deg[ei[N_EDGES + e]], w);   // col = e1 (fwd direction)
    }
}

__global__ void deg_rsqrt(float* d) {
    int i = blockIdx.x * blockDim.x + threadIdx.x;
    if (i < N_NODES) d[i] = rsqrtf(d[i]);
}

// agg = gcn_b + 2*dinv[n]^2 * m[n]   (self-loop contribution + bias)
__global__ void agg_init(const float* __restrict__ m, const float* __restrict__ dinv,
                         const float* __restrict__ bias, float* __restrict__ agg) {
    int idx = blockIdx.x * blockDim.x + threadIdx.x;
    if (idx < N_NODES * HDIM) {
        const int n = idx >> 7, f = idx & 127;
        const float di = dinv[n];
        agg[idx] = bias[f] + 2.f * di * di * m[idx];
    }
}

// One wave per undirected edge pair; both directions with one gather each.
__global__ __launch_bounds__(256) void agg_edges(
    const int* __restrict__ ei, const float* __restrict__ ewp,
    const float* __restrict__ m, const float* __restrict__ dinv,
    float* __restrict__ agg) {
    const int lane = threadIdx.x & 31;
    const int wid  = (blockIdx.x * blockDim.x + threadIdx.x) >> 5;
    const int nw   = (gridDim.x * blockDim.x) >> 5;
    const float ews = ewp[0];
    for (int e = wid; e < N_EDGES; e += nw) {
        const int a = ei[e], b = ei[N_EDGES + e];
        if (e + nw < N_EDGES) {   // L2-resident; prefetch next rows anyway
            __builtin_prefetch(m + (size_t)ei[e + nw] * HDIM, 0, 1);
            __builtin_prefetch(m + (size_t)ei[N_EDGES + e + nw] * HDIM, 0, 1);
        }
        const float w = ews * dinv[a] * dinv[b];
        const float4 va = ((const float4*)(m + (size_t)a * HDIM))[lane];
        const float4 vb = ((const float4*)(m + (size_t)b * HDIM))[lane];
        float* pa = agg + (size_t)a * HDIM + lane * 4;
        float* pb = agg + (size_t)b * HDIM + lane * 4;
        atomic_add_f32(pb + 0, w * va.x); atomic_add_f32(pb + 1, w * va.y);
        atomic_add_f32(pb + 2, w * va.z); atomic_add_f32(pb + 3, w * va.w);
        atomic_add_f32(pa + 0, w * vb.x); atomic_add_f32(pa + 1, w * vb.y);
        atomic_add_f32(pa + 2, w * vb.z); atomic_add_f32(pa + 3, w * vb.w);
    }
}

// GraphNorm stats: LDS-staged partial sums (ds_add), flushed with global atomics
__global__ __launch_bounds__(256) void gn_sum(const float* __restrict__ x,
                                              const int* __restrict__ batch,
                                              float* __restrict__ gsum) {
    __shared__ float s[N_GRAPH * HDIM];
    for (int i = threadIdx.x; i < N_GRAPH * HDIM; i += 256) s[i] = 0.f;
    __syncthreads();
    for (int idx = blockIdx.x * 256 + threadIdx.x; idx < N_NODES * HDIM;
         idx += gridDim.x * 256) {
        const int n = idx >> 7, f = idx & 127, g = batch[n];
        atomicAdd(&s[g * HDIM + f], x[idx]);
    }
    __syncthreads();
    for (int i = threadIdx.x; i < N_GRAPH * HDIM; i += 256)
        atomic_add_f32(&gsum[i], s[i]);
}

__global__ void gn_mean(const float* __restrict__ gsum, const float* __restrict__ gcnt,
                        float* __restrict__ gmean) {
    int i = blockIdx.x * blockDim.x + threadIdx.x;
    if (i < N_GRAPH * HDIM) gmean[i] = gsum[i] / gcnt[i >> 7];
}

__global__ __launch_bounds__(256) void gn_center(
    const float* __restrict__ x, const float* __restrict__ gmean,
    const float* __restrict__ ms, const int* __restrict__ batch,
    float* __restrict__ o, float* __restrict__ gvar) {
    __shared__ float s[N_GRAPH * HDIM];
    for (int i = threadIdx.x; i < N_GRAPH * HDIM; i += 256) s[i] = 0.f;
    __syncthreads();
    for (int idx = blockIdx.x * 256 + threadIdx.x; idx < N_NODES * HDIM;
         idx += gridDim.x * 256) {
        const int n = idx >> 7, f = idx & 127, g = batch[n];
        const float v = x[idx] - gmean[g * HDIM + f] * ms[f];
        o[idx] = v;
        atomicAdd(&s[g * HDIM + f], v * v);
    }
    __syncthreads();
    for (int i = threadIdx.x; i < N_GRAPH * HDIM; i += 256)
        atomic_add_f32(&gvar[i], s[i]);
}

__global__ void gn_istd(const float* __restrict__ gvar, const float* __restrict__ gcnt,
                        float* __restrict__ gistd) {
    int i = blockIdx.x * blockDim.x + threadIdx.x;
    if (i < N_GRAPH * HDIM) gistd[i] = rsqrtf(gvar[i] / gcnt[i >> 7] + GN_EPS);
}

// h = elu(o * istd * w + b) [+ skip]
__global__ void gn_apply(const float* __restrict__ o, const float* __restrict__ gistd,
                         const float* __restrict__ w, const float* __restrict__ b,
                         const int* __restrict__ batch, const float* __restrict__ skip,
                         int use_resid, float* __restrict__ hout) {
    int idx = blockIdx.x * blockDim.x + threadIdx.x;
    if (idx < N_NODES * HDIM) {
        const int n = idx >> 7, f = idx & 127, g = batch[n];
        float v = o[idx] * gistd[g * HDIM + f] * w[f] + b[f];
        v = v > 0.f ? v : (__expf(v) - 1.f);
        if (use_resid) v += skip[idx];
        hout[idx] = v;
    }
}

// final head layer: (N x 64) @ (64 x 1) + b, sigmoid
__global__ void head_final(const float* __restrict__ p2, const float* __restrict__ w3,
                           const float* __restrict__ b3, float* __restrict__ out) {
    int n = blockIdx.x * blockDim.x + threadIdx.x;
    if (n < N_NODES) {
        const float* r = p2 + (size_t)n * (HDIM / 2);
        float z = b3[0];
#pragma unroll
        for (int j = 0; j < HDIM / 2; ++j) z += r[j] * w3[j];
        out[n] = 1.f / (1.f + __expf(-z));
    }
}

// ---------------------------------------------------------------------------
extern "C" void kernel_launch(void* const* d_in, const int* in_sizes, int n_in,
                              void* d_out, int out_size, void* d_ws, size_t ws_size,
                              hipStream_t stream) {
    const float* x       = (const float*)d_in[0];
    const int*   ei      = (const int*)d_in[1];
    const int*   batch   = (const int*)d_in[2];
    const float* ewp     = (const float*)d_in[3];
    const float* ip_w1   = (const float*)d_in[4];
    const float* ip_b1   = (const float*)d_in[5];
    const float* ip_w2   = (const float*)d_in[6];
    const float* ip_b2   = (const float*)d_in[7];
    const float* gcn_w   = (const float*)d_in[8];
    const float* gcn_b   = (const float*)d_in[9];
    const float* norm_w  = (const float*)d_in[10];
    const float* norm_b  = (const float*)d_in[11];
    const float* norm_ms = (const float*)d_in[12];
    const float* skip_w  = (const float*)d_in[13];
    const float* skip_b  = (const float*)d_in[14];
    const float* ph_w1   = (const float*)d_in[15];
    const float* ph_b1   = (const float*)d_in[16];
    const float* ph_w2   = (const float*)d_in[17];
    const float* ph_b2   = (const float*)d_in[18];
    const float* ph_w3   = (const float*)d_in[19];
    const float* ph_b3   = (const float*)d_in[20];

    const int NH = N_NODES * HDIM;
    float* w = (float*)d_ws;
    float* dinv  = w; w += ((N_NODES + 255) / 256) * 256;
    float* hA    = w; w += NH;   // current h
    float* hB    = w; w += NH;   // agg accumulator
    float* mb    = w; w += NH;   // messages / centered-out / p1
    float* sb    = w; w += NH;   // skip output / p2
    float* gsum  = w; w += N_GRAPH * HDIM;
    float* gmean = w; w += N_GRAPH * HDIM;
    float* gvar  = w; w += N_GRAPH * HDIM;
    float* gistd = w; w += N_GRAPH * HDIM;
    float* gcnt  = w; w += 256;

    dim3 b256(256);
    auto g1 = [](int n) { return dim3((n + 255) / 256); };
    dim3 gemmGrid((N_NODES + 63) / 64), gemmBlk(128);

    // per-graph node counts (fixed across layers)
    zero_f32<<<g1(N_GRAPH), b256, 0, stream>>>(gcnt, N_GRAPH);
    count_nodes<<<g1(N_NODES), b256, 0, stream>>>(batch, gcnt);

    // symmetric GCN normalization: deg -> dinv (in place)
    deg_init<<<g1(N_NODES), b256, 0, stream>>>(dinv);
    deg_accum<<<g1(N_EDGES), b256, 0, stream>>>(ei, ewp, dinv);
    deg_rsqrt<<<g1(N_NODES), b256, 0, stream>>>(dinv);

    // input projection: relu(x@W1+b1)@W2+b2
    gemm_bf16_wmma<1, DIN,  HDIM><<<gemmGrid, gemmBlk, 0, stream>>>(x,  ip_w1, ip_b1, mb, N_NODES);
    gemm_bf16_wmma<0, HDIM, HDIM><<<gemmGrid, gemmBlk, 0, stream>>>(mb, ip_w2, ip_b2, hA, N_NODES);

    for (int i = 0; i < NLAYERS; ++i) {
        gemm_bf16_wmma<0, HDIM, HDIM><<<gemmGrid, gemmBlk, 0, stream>>>(
            hA, gcn_w + i * HDIM * HDIM, nullptr, mb, N_NODES);
        const int resid = (i % 2 == 0);
        if (resid)
            gemm_bf16_wmma<0, HDIM, HDIM><<<gemmGrid, gemmBlk, 0, stream>>>(
                hA, skip_w + i * HDIM * HDIM, skip_b + i * HDIM, sb, N_NODES);

        agg_init<<<g1(NH), b256, 0, stream>>>(mb, dinv, gcn_b + i * HDIM, hB);
        agg_edges<<<dim3(1024), b256, 0, stream>>>(ei, ewp, mb, dinv, hB);

        zero_f32<<<g1(N_GRAPH * HDIM), b256, 0, stream>>>(gsum, N_GRAPH * HDIM);
        gn_sum<<<dim3(480), b256, 0, stream>>>(hB, batch, gsum);
        gn_mean<<<g1(N_GRAPH * HDIM), b256, 0, stream>>>(gsum, gcnt, gmean);
        zero_f32<<<g1(N_GRAPH * HDIM), b256, 0, stream>>>(gvar, N_GRAPH * HDIM);
        gn_center<<<dim3(480), b256, 0, stream>>>(hB, gmean, norm_ms + i * HDIM, batch, mb, gvar);
        gn_istd<<<g1(N_GRAPH * HDIM), b256, 0, stream>>>(gvar, gcnt, gistd);
        gn_apply<<<g1(NH), b256, 0, stream>>>(mb, gistd, norm_w + i * HDIM,
                                              norm_b + i * HDIM, batch, sb, resid, hA);
    }

    // prediction head
    gemm_bf16_wmma<1, HDIM, HDIM><<<gemmGrid, gemmBlk, 0, stream>>>(hA, ph_w1, ph_b1, mb, N_NODES);
    gemm_bf16_wmma<1, HDIM, 64  ><<<gemmGrid, gemmBlk, 0, stream>>>(mb, ph_w2, ph_b2, sb, N_NODES);
    head_final<<<g1(N_NODES), b256, 0, stream>>>(sb, ph_w3, ph_b3, (float*)d_out);
}
